// GraphCNN_4947802325631
// MI455X (gfx1250) — matrix operand order repaired and checked by reading, not compiled, not run on hardware
//
#include <hip/hip_runtime.h>

typedef unsigned int u32;
typedef __attribute__((ext_vector_type(16))) _Float16 v16h;
typedef __attribute__((ext_vector_type(8)))  float    v8f;

#define GN   16384
#define GDIN 64
#define GH   256
#define GB   128
#define BNEPS 1e-5f

// ---------------- WMMA GEMM tiling ----------------
// Block: 256 threads = 8 waves (wave32). Block tile 64(M) x 256(N), k-chunk 32.
// Waves arranged 4(M) x 2(N); each wave computes 16x128 = 8 WMMA 16x16x32 f16 tiles.
// LDS holds k-pair-packed u32 tiles (two f16 k-values per u32) so that both the
// A fragment (pairs j<4:pk=half*4+j, j>=4:pk=8+half*4+(j-4)) and the B fragment
// (same pk order) load as aligned 16B ds reads. Strides padded to 20 u32 for
// conflict-free b128 access across the 64 LDS banks.
#define KC  32
#define TBM 64
#define TBN 256
#define AS  20
#define BS  20
#define AW  (TBM*AS)       // 1280 u32
#define BW  (TBN*BS)       // 5120 u32
#define STG (AW+BW)        // 6400 u32 per stage (double buffered: 51.2 KB LDS)

enum { AM_F16 = 0, AM_MASK = 2 };

__device__ inline u32 pack2h(float a, float b){
  union { u32 u; _Float16 h[2]; } x;
  x.h[0] = (_Float16)a; x.h[1] = (_Float16)b;
  return x.u;
}

// A fragment: 16x32 f16, lane L: row = L&15, half = L>>4.
__device__ inline v16h load_afrag(const u32* LA, int row, int half){
  union { v16h h; uint4 q[2]; } f;
  const u32* p = LA + row*AS + half*4;
  f.q[0] = *(const uint4*)p;        // VGPR j=0..3  (pk = half*4 + j)
  f.q[1] = *(const uint4*)(p + 8);  // VGPR j=4..7  (pk = 8 + half*4 + j-4)
  return f.h;
}

// B fragment: 32x16 f16, lane L: col = L&15, half = L>>4, VGPR v<4: pk = half*4 + v, v>=4: pk = 8 + half*4 + (v-4).
__device__ inline v16h load_bfrag(const u32* LB, int c, int half){
  union { v16h h; uint4 q[2]; } f;
  const u32* p = LB + c*BS + half*4;
  f.q[0] = *(const uint4*)p;
  f.q[1] = *(const uint4*)(p + 8);
  return f.h;
}

// Expand one 32-bit adjacency mask word (k-chunk of one row) into an exact
// f16 0/1 A fragment for this lane's half.
__device__ inline v16h mask_afrag(u32 m, int half){
  union { v16h h; u32 u[8]; } f;
#pragma unroll
  for (int j = 0; j < 8; ++j){
    int k0 = ((j & 4) << 2) + half*8 + ((j & 3) << 1);  // (j>=4?16:0)+half*8+2*(j&3)
    f.u[j] = (((m >> k0) & 1u)     ? 0x00003C00u : 0u)   // f16 1.0 (lo)
           | (((m >> (k0+1)) & 1u) ? 0x3C000000u : 0u);  // f16 1.0 (hi)
  }
  return f.h;
}

// ---- CDNA5 async global->LDS staging of the A tile (ASYNCcnt-tracked) ------
__device__ inline u32 lds_off32(const u32* p){        // low 32 bits = LDS byte addr
  return (u32)(size_t)p;
}
__device__ inline void async_a128(const u32* lds_dst, const u32* g){
  asm volatile("global_load_async_to_lds_b128 %0, %1, off"
               :: "v"(lds_off32(lds_dst)), "v"(g) : "memory");
}
__device__ inline void async_a32(const u32* lds_dst, const u32* g){
  asm volatile("global_load_async_to_lds_b32 %0, %1, off"
               :: "v"(lds_off32(lds_dst)), "v"(g) : "memory");
}
__device__ inline void wait_async0(){
  asm volatile("s_wait_asynccnt 0x0" ::: "memory");
}

template<int AMODE>
__device__ inline void issueA_async(u32* LA, const u32* __restrict__ Ap,
                                    const u32* __restrict__ Amask,
                                    int mb, int kc, int K, int tid)
{
  if (AMODE == AM_MASK){
    if (tid < TBM)   // one mask word per row of the block tile
      async_a32(&LA[tid], &Amask[(size_t)(mb + tid) * (size_t)(K >> 5) + (kc >> 5)]);
  } else {
    // A is row-major f16 => u32 view is already k-pair packed; 16B chunks are
    // contiguous in both global memory and (padded) LDS.
    int row = tid >> 2, pk0 = (tid & 3) << 2;
    async_a128(&LA[row*AS + pk0],
               &Ap[(size_t)(mb + row) * (size_t)(K >> 1) + (kc >> 1) + pk0]);
  }
}

// ---- register-staged B tile (transposed into LDS at store time) ------------
__device__ inline void loadB_regs(uint4 v[4], const u32* __restrict__ Bp,
                                  int kc, int Nc, int tid)
{
  int pkg = tid >> 6, c0 = (tid & 63) << 2;
#pragma unroll
  for (int i = 0; i < 4; ++i){
    int pk = pkg*4 + i;
    if (c0 < Nc) v[i] = *(const uint4*)&Bp[(size_t)((kc >> 1) + pk) * Nc + c0];
    else         { v[i].x = 0; v[i].y = 0; v[i].z = 0; v[i].w = 0; }
  }
}
__device__ inline void storeB_lds(const uint4 v[4], u32* LB, int tid)
{
  int pkg = tid >> 6, c0 = (tid & 63) << 2;
#pragma unroll
  for (int i = 0; i < 4; ++i){
    int pk = pkg*4 + i;
    LB[(c0+0)*BS + pk] = v[i].x;
    LB[(c0+1)*BS + pk] = v[i].y;
    LB[(c0+2)*BS + pk] = v[i].z;
    LB[(c0+3)*BS + pk] = v[i].w;
  }
}

template<int AMODE>
__global__ __launch_bounds__(256)
void gemm_wmma(const u32* __restrict__ Ap, const u32* __restrict__ Amask,
               const u32* __restrict__ Bp,
               float* __restrict__ C32, _Float16* __restrict__ C16,
               const float* __restrict__ bias,
               int M, int Nc, int K)
{
  __shared__ u32 lds[2*STG];
  const int tid   = threadIdx.x;
  const int lane  = tid & 31, wave = tid >> 5;
  const int waveM = wave >> 1, waveN = wave & 1;
  const int r     = lane & 15, half = lane >> 4;
  const int mb    = blockIdx.x * TBM;

  v8f acc[8] = {};
  uint4 breg[4];

  const int nk = K / KC;
  // prologue: issue tile 0 (A async into buf0, B into registers)
  loadB_regs(breg, Bp, 0, Nc, tid);
  issueA_async<AMODE>(lds, Ap, Amask, mb, 0, K, tid);

  for (int t = 0; t < nk; ++t){
    const int cur = t & 1;
    u32* LAc = lds + cur*STG;
    u32* LBc = LAc + AW;

    // Make LDS[cur] ready: store register-staged B (compiler waits loadcnt
    // here, a full compute block after the loads were issued), drain the
    // in-order async A copies for this buffer, then rendezvous.
    storeB_lds(breg, LBc, tid);
    wait_async0();
    __syncthreads();

    // Fire-and-forget staging for tile t+1 into the other buffer.
    if (t + 1 < nk){
      loadB_regs(breg, Bp, (t+1)*KC, Nc, tid);
      issueA_async<AMODE>(lds + (cur^1)*STG, Ap, Amask, mb, (t+1)*KC, K, tid);
    }

    // Compute: 8 WMMAs per wave, latency of the staged loads hides behind them.
    v16h a = (AMODE == AM_MASK) ? mask_afrag(LAc[waveM*16 + r], half)
                                : load_afrag(LAc, waveM*16 + r, half);
#pragma unroll
    for (int s = 0; s < 8; ++s){
      v16h b = load_bfrag(LBc, waveN*128 + s*16 + r, half);
      acc[s] = __builtin_amdgcn_wmma_f32_16x16x32_f16(
                   false, a, false, b, (short)0, acc[s], false, false);
    }
    // no trailing barrier needed: next iteration's LDS writes target the other
    // buffer, and its pre-compute barrier orders everything.
  }

  // Epilogue: C VGPR i -> row = i + 8*half, col = lane&15 within 16x16 tile.
#pragma unroll
  for (int s = 0; s < 8; ++s){
    int col = waveN*128 + s*16 + r;
    if (col < Nc){
      float bb = bias ? bias[col] : 0.0f;
#pragma unroll
      for (int i = 0; i < 8; ++i){
        int row = mb + waveM*16 + half*8 + i;
        float v = acc[s][i] + bb;
        if (C32) C32[(size_t)row*Nc + col] = v;
        if (C16) C16[(size_t)row*Nc + col] = (_Float16)v;
      }
    }
  }
}

// ---------------- adjacency -> bitmask (1 bit per entry, exact 0/1) ----------
__global__ __launch_bounds__(256)
void build_mask_kernel(const float* __restrict__ adj, u32* __restrict__ mask)
{
  size_t wv = (size_t)blockIdx.x * 256 + threadIdx.x;   // one u32 word / thread
  const float* p = adj + wv*32;
  u32 m = 0;
#pragma unroll
  for (int i = 0; i < 32; ++i) m |= (p[i] != 0.0f) ? (1u << i) : 0u;
  mask[wv] = m;
}

// ---------------- f32 [K,Nc] -> row-pair packed f16 u32 [K/2, Nc] -------------
__global__ __launch_bounds__(256)
void pack_kernel(const float* __restrict__ in, u32* __restrict__ outp, int K, int Nc)
{
  int idx = blockIdx.x*256 + threadIdx.x;
  if (idx >= (K/2)*Nc) return;
  int pk = idx / Nc, c = idx % Nc;
  outp[idx] = pack2h(in[(size_t)(2*pk)*Nc + c], in[(size_t)(2*pk+1)*Nc + c]);
}

// ---------------- column stats (sum, sumsq) over [M, 256] --------------------
__global__ __launch_bounds__(256)
void colstats_kernel(const float* __restrict__ t, int M,
                     float* __restrict__ sum, float* __restrict__ sq)
{
  int c = threadIdx.x;
  size_t r0 = (size_t)blockIdx.x * 128;
  float s = 0.f, q = 0.f;
  for (int i = 0; i < 128; ++i){
    float v = t[(r0 + i)*GH + c];
    s += v; q += v*v;
  }
  atomicAdd(&sum[c], s);
  atomicAdd(&sq[c],  q);
}

// ---------------- BN(train, biased var) + ReLU + repack ----------------------
__global__ __launch_bounds__(256)
void bn_relu_kernel(const float* __restrict__ t,
                    const float* __restrict__ sum, const float* __restrict__ sq,
                    const float* __restrict__ gamma, const float* __restrict__ beta,
                    int M, _Float16* __restrict__ out16, u32* __restrict__ outp,
                    float* __restrict__ out32)
{
  int c  = threadIdx.x;
  int pk = blockIdx.x;                 // row pair
  float inv  = 1.0f / (float)M;
  float mean = sum[c] * inv;
  float var  = fmaxf(sq[c]*inv - mean*mean, 0.0f);
  float g    = gamma[c] * rsqrtf(var + BNEPS);
  float sh   = beta[c] - g*mean;
  size_t i0  = (size_t)(2*pk)*GH + c;
  float v0 = fmaxf(t[i0]      * g + sh, 0.0f);
  float v1 = fmaxf(t[i0 + GH] * g + sh, 0.0f);
  if (out16){ out16[i0] = (_Float16)v0; out16[i0 + GH] = (_Float16)v1; }
  if (outp ){ outp[(size_t)pk*GH + c] = pack2h(v0, v1); }
  if (out32){ out32[i0] = v0; out32[i0 + GH] = v1; }
}

// ---------------- readout: graph_pool(f32) @ h2(f16 packed), split-K ---------
__global__ __launch_bounds__(256)
void pool_kernel(const float* __restrict__ gp, const u32* __restrict__ h2p,
                 float* __restrict__ out)
{
  int b = blockIdx.x, ks = blockIdx.y, c = threadIdx.x;
  const int PKS = (GN/2)/8;                 // 1024 row-pairs per k-slice
  int pk0 = ks*PKS;
  const float* gpr = gp + (size_t)b*GN;
  float s = 0.f;
  for (int pk = pk0; pk < pk0 + PKS; ++pk){
    union { u32 u; _Float16 h[2]; } v;
    v.u = h2p[(size_t)pk*GH + c];
    s += gpr[2*pk] * (float)v.h[0] + gpr[2*pk+1] * (float)v.h[1];
  }
  atomicAdd(&out[(size_t)b*GH + c], s);
}

extern "C" void kernel_launch(void* const* d_in, const int* in_sizes, int n_in,
                              void* d_out, int out_size, void* d_ws, size_t ws_size,
                              hipStream_t stream)
{
  (void)in_sizes; (void)n_in; (void)out_size; (void)ws_size;
  const float* x    = (const float*)d_in[0];
  const float* adj  = (const float*)d_in[1];
  const float* gp   = (const float*)d_in[2];
  const float* w00  = (const float*)d_in[3];
  const float* b00  = (const float*)d_in[4];
  const float* g00  = (const float*)d_in[5];
  const float* be00 = (const float*)d_in[6];
  const float* w01  = (const float*)d_in[7];
  const float* b01  = (const float*)d_in[8];
  const float* g0   = (const float*)d_in[9];
  const float* be0  = (const float*)d_in[10];
  const float* w10  = (const float*)d_in[11];
  const float* b10  = (const float*)d_in[12];
  const float* g10  = (const float*)d_in[13];
  const float* be10 = (const float*)d_in[14];
  const float* w11  = (const float*)d_in[15];
  const float* b11  = (const float*)d_in[16];
  const float* g1   = (const float*)d_in[17];
  const float* be1  = (const float*)d_in[18];
  float* out = (float*)d_out;

  // ---- workspace layout (256B aligned slices) ----
  char* wp = (char*)d_ws;
  auto alloc = [&](size_t bytes)->char*{
    char* p = wp; wp += (bytes + 255) & ~(size_t)255; return p;
  };
  u32*      maskb   = (u32*)     alloc((size_t)GN*(GN/32)*4);   // 32 MB
  u32*      xp      = (u32*)     alloc((size_t)(GN/2)*GDIN*4);  // x packed
  u32*      w00p    = (u32*)     alloc((size_t)(GDIN/2)*GH*4);
  u32*      w01p    = (u32*)     alloc((size_t)(GH/2)*GH*4);
  u32*      w10p    = (u32*)     alloc((size_t)(GH/2)*GH*4);
  u32*      w11p    = (u32*)     alloc((size_t)(GH/2)*GH*4);
  _Float16* pooled1 = (_Float16*)alloc((size_t)GN*GDIN*2);
  _Float16* pooled2 = (_Float16*)alloc((size_t)GN*GH*2);
  float*    t       = (float*)   alloc((size_t)GN*GH*4);        // pre-BN f32
  _Float16* r16     = (_Float16*)alloc((size_t)GN*GH*2);        // A-side f16
  u32*      h1p     = (u32*)     alloc((size_t)(GN/2)*GH*4);    // B-side packed
  u32*      h2p     = (u32*)     alloc((size_t)(GN/2)*GH*4);
  float*    ssum    = (float*)   alloc(GH*4);
  float*    ssq     = (float*)   alloc(GH*4);

  auto bn = [&](const float* gamma, const float* beta,
                _Float16* o16, u32* opk, float* o32){
    hipMemsetAsync(ssum, 0, GH*4, stream);
    hipMemsetAsync(ssq,  0, GH*4, stream);
    colstats_kernel<<<GN/128, 256, 0, stream>>>(t, GN, ssum, ssq);
    bn_relu_kernel<<<GN/2, 256, 0, stream>>>(t, ssum, ssq, gamma, beta, GN,
                                             o16, opk, o32);
  };

  // 0) compress adjacency to bits; pack x and weights to f16 pairs
  build_mask_kernel<<<(GN*(GN/32))/256, 256, 0, stream>>>(adj, maskb);
  pack_kernel<<<((GN/2)*GDIN + 255)/256, 256, 0, stream>>>(x,   xp,   GN,   GDIN);
  pack_kernel<<<((GDIN/2)*GH  + 255)/256, 256, 0, stream>>>(w00, w00p, GDIN, GH);
  pack_kernel<<<((GH/2)*GH    + 255)/256, 256, 0, stream>>>(w01, w01p, GH,   GH);
  pack_kernel<<<((GH/2)*GH    + 255)/256, 256, 0, stream>>>(w10, w10p, GH,   GH);
  pack_kernel<<<((GH/2)*GH    + 255)/256, 256, 0, stream>>>(w11, w11p, GH,   GH);

  // ---- layer 1 ----
  gemm_wmma<AM_MASK><<<GN/TBM, 256, 0, stream>>>(
      nullptr, maskb, xp, nullptr, pooled1, nullptr, GN, GDIN, GN);
  gemm_wmma<AM_F16><<<GN/TBM, 256, 0, stream>>>(
      (const u32*)pooled1, nullptr, w00p, t, nullptr, b00, GN, GH, GDIN);
  bn(g00, be00, r16, nullptr, nullptr);
  gemm_wmma<AM_F16><<<GN/TBM, 256, 0, stream>>>(
      (const u32*)r16, nullptr, w01p, t, nullptr, b01, GN, GH, GH);
  bn(g0, be0, nullptr, h1p, nullptr);

  // ---- layer 2 ----
  gemm_wmma<AM_MASK><<<GN/TBM, 256, 0, stream>>>(
      nullptr, maskb, h1p, nullptr, pooled2, nullptr, GN, GH, GN);
  gemm_wmma<AM_F16><<<GN/TBM, 256, 0, stream>>>(
      (const u32*)pooled2, nullptr, w10p, t, nullptr, b10, GN, GH, GH);
  bn(g10, be10, r16, nullptr, nullptr);
  gemm_wmma<AM_F16><<<GN/TBM, 256, 0, stream>>>(
      (const u32*)r16, nullptr, w11p, t, nullptr, b11, GN, GH, GH);
  // h_nodes (f32) straight into d_out, plus packed copy for the readout GEMM
  bn(g1, be1, nullptr, h2p, out + (size_t)GB*GH);

  // ---- graph readout: pooled_h = graph_pool @ h2 ----
  hipMemsetAsync(out, 0, (size_t)GB*GH*4, stream);
  pool_kernel<<<dim3(GB, 8), 256, 0, stream>>>(gp, h2p, out);
}
